// SAGE_7516192768895
// MI455X (gfx1250) — matrix operand order repaired and checked by reading, not compiled, not run on hardware
//
#include <hip/hip_runtime.h>

#define HID 128
#define TILE_M 64
#define WSTR 264   // padded LDS row stride (bf16 elems) for weights (K=256)
#define ASTR 264   // padded LDS row stride (bf16 elems) for A tile (K=256)
#define GEMM_GRID 512

typedef __attribute__((ext_vector_type(16))) __bf16 v16bf;
typedef __attribute__((ext_vector_type(8)))  __bf16 v8bf;
typedef __attribute__((ext_vector_type(4)))  __bf16 v4bf;
typedef __attribute__((ext_vector_type(8)))  float  v8f;

// Build a 16-elem bf16 fragment from two 16-byte LDS chunks.
__device__ __forceinline__ v16bf ld_frag2(const __bf16* p0, const __bf16* p1) {
  v8bf lo = *(const v8bf*)p0;
  v8bf hi = *(const v8bf*)p1;
  v16bf r;
#pragma unroll
  for (int i = 0; i < 8; ++i) { r[i] = lo[i]; r[i + 8] = hi[i]; }
  return r;
}

__global__ void zero_i32_kernel(int* __restrict__ p, int n) {
  int i = blockIdx.x * blockDim.x + threadIdx.x;
  if (i < n) p[i] = 0;
}

__global__ void hist_kernel(const int* __restrict__ dst, int* __restrict__ cnt, int nE) {
  int e = blockIdx.x * blockDim.x + threadIdx.x;
  if (e < nE) atomicAdd(&cnt[dst[e]], 1);
}

// Single-workgroup chunked Hillis-Steele exclusive scan: rowptr[i] = sum cnt[0..i)
__global__ __launch_bounds__(1024) void scan_kernel(const int* __restrict__ cnt,
                                                    int* __restrict__ rowptr, int n) {
  __shared__ int buf[1024];
  __shared__ int carry;
  int t = threadIdx.x;
  if (t == 0) carry = 0;
  __syncthreads();
  for (int base = 0; base < n; base += 1024) {
    int v = (base + t < n) ? cnt[base + t] : 0;
    buf[t] = v;
    __syncthreads();
#pragma unroll
    for (int off = 1; off < 1024; off <<= 1) {
      int x = (t >= off) ? buf[t - off] : 0;
      __syncthreads();
      buf[t] += x;
      __syncthreads();
    }
    int incl = buf[t];
    if (base + t < n) rowptr[base + t] = carry + incl - v;
    __syncthreads();
    if (t == 0) carry += buf[1023];
    __syncthreads();
  }
  if (t == 0) rowptr[n] = carry;
}

__global__ void copy_i32_kernel(const int* __restrict__ a, int* __restrict__ b, int n) {
  int i = blockIdx.x * blockDim.x + threadIdx.x;
  if (i < n) b[i] = a[i];
}

// csr[slot] = src[e], slots grouped by dst via running cursor (= rowptr copy)
__global__ void fill_kernel(const int* __restrict__ src, const int* __restrict__ dst,
                            int* __restrict__ cursor, int* __restrict__ csr, int nE) {
  int e = blockIdx.x * blockDim.x + threadIdx.x;
  if (e >= nE) return;
  int slot = atomicAdd(&cursor[dst[e]], 1);
  csr[slot] = src[e];
}

// x[node] = bf16(z_emb[z[node]])  (lane covers 4 feats; 32 lanes = 128)
__global__ void embed_kernel(const int* __restrict__ z, const float* __restrict__ z_emb,
                             __bf16* __restrict__ x, int nNodes) {
  int gid = blockIdx.x * blockDim.x + threadIdx.x;
  int node = gid >> 5, lane = gid & 31;
  if (node >= nNodes) return;
  int zi = z[node];
  float4 v = *(const float4*)(z_emb + (size_t)zi * HID + lane * 4);
  v4bf o = {(__bf16)v.x, (__bf16)v.y, (__bf16)v.z, (__bf16)v.w};
  *(v4bf*)(x + (size_t)node * HID + lane * 4) = o;
}

// Gather-reduce mean aggregation: one wave per node, f32 accumulate, bf16 storage.
__global__ __launch_bounds__(256) void aggregate_kernel(
    const __bf16* __restrict__ x, const int* __restrict__ rowptr,
    const int* __restrict__ csr, __bf16* __restrict__ agg, int nNodes) {
  int gid = blockIdx.x * blockDim.x + threadIdx.x;
  int node = gid >> 5, lane = gid & 31;
  if (node >= nNodes) return;
  int beg = rowptr[node], end = rowptr[node + 1];
  float a0 = 0.f, a1 = 0.f, a2 = 0.f, a3 = 0.f;
  for (int j = beg; j < end; j += 32) {
    int idx = (j + lane < end) ? csr[j + lane] : 0;
    int m = end - j;
    if (m > 32) m = 32;
    int i = 0;
    for (; i + 1 < m; i += 2) {  // 2 outstanding gathers per iter
      int s0 = __shfl(idx, i, 32);
      int s1 = __shfl(idx, i + 1, 32);
      v4bf u = *(const v4bf*)(x + (size_t)s0 * HID + lane * 4);
      v4bf w = *(const v4bf*)(x + (size_t)s1 * HID + lane * 4);
      a0 += (float)u[0] + (float)w[0];
      a1 += (float)u[1] + (float)w[1];
      a2 += (float)u[2] + (float)w[2];
      a3 += (float)u[3] + (float)w[3];
    }
    if (i < m) {
      int s0 = __shfl(idx, i, 32);
      v4bf u = *(const v4bf*)(x + (size_t)s0 * HID + lane * 4);
      a0 += (float)u[0]; a1 += (float)u[1]; a2 += (float)u[2]; a3 += (float)u[3];
    }
  }
  float inv = 1.0f / fmaxf((float)(end - beg), 1.0f);
  v4bf r = {(__bf16)(a0 * inv), (__bf16)(a1 * inv), (__bf16)(a2 * inv), (__bf16)(a3 * inv)};
  *(v4bf*)(agg + (size_t)node * HID + lane * 4) = r;
}

// out = maybe_relu( [agg | x] @ [Wl ; Wr]^T + b )  via bf16 WMMA, f32 acc.
// Grid-stride over M tiles: weights staged in LDS once per block.
__global__ __launch_bounds__(256) void sage_gemm_kernel(
    const __bf16* __restrict__ agg, const __bf16* __restrict__ xin,
    const float* __restrict__ Wl, const float* __restrict__ Wr,
    const float* __restrict__ bias, __bf16* __restrict__ xout,
    int nNodes, int do_relu) {
  __shared__ __align__(16) __bf16 Wlds[HID * WSTR];     // 128 rows x 256 K
  __shared__ __align__(16) __bf16 Alds[TILE_M * ASTR];  // 64 rows x 256 K

  int t = threadIdx.x;

  // Stage combined weight [n][k]: k<128 -> Wl[n][k], else Wr[n][k-128]
  for (int f = t; f < HID * 256; f += 256) {
    int row = f >> 8, col = f & 255;
    float v = (col < HID) ? Wl[row * HID + col] : Wr[row * HID + (col - HID)];
    Wlds[row * WSTR + col] = (__bf16)v;
  }

  int lane = t & 31;
  int wave = t >> 5;
  int wm = wave & 3;    // 16-row M block within tile
  int wn = wave >> 2;   // 64-col N half
  int r = lane & 15;
  int hsel = lane >> 4; // lane half

  int numTiles = (nNodes + TILE_M - 1) / TILE_M;
  for (int tile = blockIdx.x; tile < numTiles; tile += gridDim.x) {
    int m0 = tile * TILE_M;

    // Stage A tile [m][k] as raw bf16 16B chunks: k<128 -> agg, else xin
    for (int f = t; f < TILE_M * 32; f += 256) {   // 32 chunks of 8 elems per row
      int row = f >> 5;
      int coloff = (f & 31) * 8;
      int gr = m0 + row;
      uint4 d = {0u, 0u, 0u, 0u};
      if (gr < nNodes) {
        const __bf16* srcp = (coloff < HID)
            ? (agg + (size_t)gr * HID + coloff)
            : (xin + (size_t)gr * HID + (coloff - HID));
        d = *(const uint4*)srcp;
      }
      *(uint4*)(Alds + row * ASTR + coloff) = d;
    }
    __syncthreads();

    v8f acc[4] = {};
    const __bf16* arow = Alds + (wm * 16 + r) * ASTR;

#pragma unroll
    for (int kb = 0; kb < 8; ++kb) {
      int k0 = kb * 32;
      // A 16x32 frag: lanes 0-15 K={0..7,16..23}, lanes 16-31 K={8..15,24..31}
      v16bf a = ld_frag2(arow + k0 + hsel * 8, arow + k0 + 16 + hsel * 8);
#pragma unroll
      for (int nt = 0; nt < 4; ++nt) {
        int n = wn * 64 + nt * 16 + r;
        // B 32x16 frag: lanes 0-15 K={0..7,16..23}, lanes 16-31 K={8..15,24..31}; B[k][n] = Wcat[n][k]
        const __bf16* brow = Wlds + n * WSTR + k0 + hsel * 8;
        v16bf b = ld_frag2(brow, brow + 16);
        acc[nt] = __builtin_amdgcn_wmma_f32_16x16x32_bf16(
            false, a, false, b, (short)0, acc[nt], false, false);
      }
    }

    // C/D layout: VGPR j -> M = j (lanes 0-15) / j+8 (lanes 16-31); N = lane&15
    int mbase = m0 + wm * 16 + hsel * 8;
#pragma unroll
    for (int nt = 0; nt < 4; ++nt) {
      int n = wn * 64 + nt * 16 + r;
      float bv = bias[n];
#pragma unroll
      for (int j = 0; j < 8; ++j) {
        int gm = mbase + j;
        if (gm < nNodes) {
          float v = acc[nt][j] + bv;
          if (do_relu) v = fmaxf(v, 0.0f);
          xout[(size_t)gm * HID + n] = (__bf16)v;
        }
      }
    }
    __syncthreads();  // protect Alds before next tile's staging
  }
}

// Per-graph: searchsorted + h = x[c]*x[c+1]; relu(h@W1^T+b1)@W2^T+b2
__global__ __launch_bounds__(128) void readout_kernel(
    const __bf16* __restrict__ x, const int* __restrict__ batch,
    const float* __restrict__ l1w, const float* __restrict__ l1b,
    const float* __restrict__ l2w, const float* __restrict__ l2b,
    float* __restrict__ out, int nNodes) {
  int g = blockIdx.x;
  int t = threadIdx.x;
  __shared__ float h[HID];
  __shared__ float partial[4];

  int lo = 0, hi = nNodes;
  while (lo < hi) {
    int mid = (lo + hi) >> 1;
    if (batch[mid] < g) lo = mid + 1; else hi = mid;
  }
  int c = lo;
  int c1 = (c + 1 < nNodes) ? (c + 1) : (nNodes - 1);
  h[t] = (float)x[(size_t)c * HID + t] * (float)x[(size_t)c1 * HID + t];
  __syncthreads();

  float acc = l1b[t];
#pragma unroll 4
  for (int k = 0; k < HID; ++k) acc += h[k] * l1w[t * HID + k];
  acc = fmaxf(acc, 0.0f);

  float p = acc * l2w[t];
#pragma unroll
  for (int off = 16; off > 0; off >>= 1) p += __shfl_down(p, off, 32);
  if ((t & 31) == 0) partial[t >> 5] = p;
  __syncthreads();
  if (t == 0) out[g] = partial[0] + partial[1] + partial[2] + partial[3] + l2b[0];
}

extern "C" void kernel_launch(void* const* d_in, const int* in_sizes, int n_in,
                              void* d_out, int out_size, void* d_ws, size_t ws_size,
                              hipStream_t stream) {
  const int*   z     = (const int*)d_in[0];
  const int*   edge  = (const int*)d_in[1];
  const int*   batch = (const int*)d_in[2];
  const float* z_emb = (const float*)d_in[4];
  const float* Wl    = (const float*)d_in[5];
  const float* bl    = (const float*)d_in[6];
  const float* Wr    = (const float*)d_in[7];
  const float* l1w   = (const float*)d_in[8];
  const float* l1b   = (const float*)d_in[9];
  const float* l2w   = (const float*)d_in[10];
  const float* l2b   = (const float*)d_in[11];
  float* out = (float*)d_out;

  int nNodes = in_sizes[0];
  int nEdges = in_sizes[1] / 2;
  int nGraphs = out_size;
  const int* src = edge;
  const int* dst = edge + nEdges;

  // Workspace layout (256B aligned): cnt | rowptr | cursor | csr | x0 | x1 | agg (bf16)
  char* ws = (char*)d_ws;
  size_t nB    = (((size_t)nNodes * sizeof(int)) + 255) & ~(size_t)255;
  size_t n1B   = (((size_t)(nNodes + 1) * sizeof(int)) + 255) & ~(size_t)255;
  size_t eB    = (((size_t)nEdges * sizeof(int)) + 255) & ~(size_t)255;
  size_t featB = (((size_t)nNodes * HID * sizeof(__bf16)) + 255) & ~(size_t)255;
  int*    cnt    = (int*)ws;
  int*    rowptr = (int*)(ws + nB);
  int*    cursor = (int*)(ws + nB + n1B);
  int*    csr    = (int*)(ws + nB + n1B + nB);
  __bf16* x0     = (__bf16*)(ws + nB + n1B + nB + eB);
  __bf16* x1     = (__bf16*)(ws + nB + n1B + nB + eB + featB);
  __bf16* agg    = (__bf16*)(ws + nB + n1B + nB + eB + 2 * featB);
  (void)ws_size; (void)n_in;

  // Build CSR (by destination) once; reused by all 3 layers.
  zero_i32_kernel<<<(nNodes + 255) / 256, 256, 0, stream>>>(cnt, nNodes);
  hist_kernel<<<(nEdges + 255) / 256, 256, 0, stream>>>(dst, cnt, nEdges);
  scan_kernel<<<1, 1024, 0, stream>>>(cnt, rowptr, nNodes);
  copy_i32_kernel<<<(nNodes + 255) / 256, 256, 0, stream>>>(rowptr, cursor, nNodes);
  fill_kernel<<<(nEdges + 255) / 256, 256, 0, stream>>>(src, dst, cursor, csr, nEdges);

  embed_kernel<<<(nNodes * 32 + 255) / 256, 256, 0, stream>>>(z, z_emb, x0, nNodes);

  int numTiles = (nNodes + TILE_M - 1) / TILE_M;
  int gemmGrid = numTiles < GEMM_GRID ? numTiles : GEMM_GRID;

  __bf16* xin = x0;
  __bf16* xout = x1;
  for (int layer = 0; layer < 3; ++layer) {
    aggregate_kernel<<<(nNodes * 32 + 255) / 256, 256, 0, stream>>>(
        xin, rowptr, csr, agg, nNodes);
    sage_gemm_kernel<<<gemmGrid, 256, 0, stream>>>(
        agg, xin,
        Wl + (size_t)layer * HID * HID, Wr + (size_t)layer * HID * HID,
        bl + (size_t)layer * HID, xout, nNodes, (layer < 2) ? 1 : 0);
    __bf16* tmp = xin; xin = xout; xout = tmp;
  }

  readout_kernel<<<nGraphs, HID, 0, stream>>>(xin, batch, l1w, l1b, l2w, l2b, out, nNodes);
}
